// Attention_48687749267827
// MI455X (gfx1250) — matrix operand order repaired and checked by reading, not compiled, not run on hardware
//
#include <hip/hip_runtime.h>

typedef __attribute__((ext_vector_type(16))) __bf16 v16bf;
typedef __attribute__((ext_vector_type(8)))  __bf16 bf16x8;
typedef __attribute__((ext_vector_type(8)))  float  v8f;

#define NB    16
#define CIN   384
#define MID   384
#define NHEAD 12
#define HDIM  32
#define NPIX  784            // 28*28 = 49 * 16 exactly
#define NKPAD 800            // keys padded to 25 * 32
#define ATTN_SCALE 0.17677669529663687f   // 32^-0.5

static __device__ __forceinline__ __bf16 tobf(float x) { return (__bf16)x; }
static __device__ __forceinline__ bf16x8 ldb8(const __bf16* p) {
  return *(const bf16x8*)p;          // 16-byte global/LDS load
}
static __device__ __forceinline__ v16bf cat16(bf16x8 lo, bf16x8 hi) {
  return __builtin_shufflevector(lo, hi, 0,1,2,3,4,5,6,7,8,9,10,11,12,13,14,15);
}

#define WMMA_BF16(A, B, C) \
  __builtin_amdgcn_wmma_f32_16x16x32_bf16(false, (A), false, (B), (short)0, (C), false, false)

// ---------------------------------------------------------------------------
// Prep 0: f32 -> bf16 weight conversion (one pass, removes converts from GEMMs)
// ---------------------------------------------------------------------------
__global__ __launch_bounds__(256) void cvt_kernel(const float* __restrict__ src,
                                                  __bf16* __restrict__ dst, int n)
{
  int i = blockIdx.x * 256 + threadIdx.x;
  if (i < n) dst[i] = tobf(src[i]);
}

// ---------------------------------------------------------------------------
// Prep 1: x [b][c][n] f32  ->  xT [b][n][c] bf16  (LDS 16x16 tile transpose)
// ---------------------------------------------------------------------------
__global__ __launch_bounds__(256) void xpose_kernel(const float* __restrict__ x,
                                                    __bf16* __restrict__ xT)
{
  __shared__ float sm[16][17];
  const int b  = blockIdx.z;
  const int c0 = blockIdx.y * 16;
  const int n0 = blockIdx.x * 16;
  const int tr = threadIdx.x >> 4;
  const int tc = threadIdx.x & 15;
  sm[tr][tc] = x[((size_t)b * CIN + c0 + tr) * NPIX + n0 + tc];
  __syncthreads();
  xT[((size_t)b * NPIX + n0 + tr) * CIN + c0 + tc] = tobf(sm[tc][tr]);
}

// ---------------------------------------------------------------------------
// Prep 2: zero the key/value padding (rows/cols 784..799)
// ---------------------------------------------------------------------------
__global__ __launch_bounds__(256) void pad_zero_kernel(__bf16* __restrict__ kpad,
                                                       __bf16* __restrict__ vT)
{
  int t  = blockIdx.x * 256 + threadIdx.x;   // 192 (b,h) pairs * 512 pad elems
  int bh = t >> 9;
  int r  = t & 511;
  if (bh < NB * NHEAD) {
    const int row = NPIX + (r >> 5), d = r & 31;
    kpad[((size_t)bh * NKPAD + row) * HDIM + d] = (__bf16)0.f;
    const int dd = r >> 4, cc = NPIX + (r & 15);
    vT[((size_t)bh * HDIM + dd) * NKPAD + cc] = (__bf16)0.f;
  }
}

// ---------------------------------------------------------------------------
// Stage 1: fused QKV GEMM.  Y = [Wq;Wkv] (1152x384, bf16) * X_b (384x784)
// 64x16 output tile per wave: 4 accumulators share one B-fragment per K-step
// (10 b128 loads per 4 WMMAs). 1152 = 18 * 64; q/k/v boundaries are 64-aligned.
// ---------------------------------------------------------------------------
__global__ __launch_bounds__(64) void qkv_kernel(
    const __bf16* __restrict__ xT, const __bf16* __restrict__ wqkv,
    const float* __restrict__ bq,  const float* __restrict__ bkv,
    __bf16* __restrict__ qw, __bf16* __restrict__ kpad, __bf16* __restrict__ vT)
{
  const int lane  = threadIdx.x & 31;
  const int wv    = threadIdx.x >> 5;
  const int mtile = blockIdx.y * 2 + wv;      // 0..17 (18 * 64 = 1152)
  const int m0    = mtile * 64;
  const int n0    = blockIdx.x * 16;          // 0..48
  const int b     = blockIdx.z;
  const int hi8   = (lane >= 16) ? 8 : 0;
  const int col   = n0 + (lane & 15);
  const int kbA   = (lane >= 16) ? 16 : 0;
  const int kbB   = (lane >= 16) ? 16 : 0;

  const __bf16* wrow0 = wqkv + (size_t)(m0 + (lane & 15)) * CIN;
  const __bf16* xrow  = xT + ((size_t)b * NPIX + col) * CIN;

  v8f acc[4];
#pragma unroll
  for (int c = 0; c < 4; ++c) acc[c] = (v8f){0.f,0.f,0.f,0.f,0.f,0.f,0.f,0.f};

  for (int k0 = 0; k0 < CIN; k0 += 32) {
    const v16bf bm = cat16(ldb8(xrow + k0 + kbB), ldb8(xrow + k0 + kbB + 8));
#pragma unroll
    for (int c = 0; c < 4; ++c) {
      const __bf16* wr = wrow0 + (size_t)(16 * c) * CIN + k0 + kbA;
      const v16bf a = cat16(ldb8(wr), ldb8(wr + 8));
      acc[c] = WMMA_BF16(a, bm, acc[c]);
    }
  }

#pragma unroll
  for (int c = 0; c < 4; ++c) {
    // 8 accumulator rows = 8 consecutive m -> 8 consecutive d within one head
    const int mbase = m0 + 16 * c + hi8;
    if (m0 < MID) {                                        // ---- q ----
      bf16x8 pk;
#pragma unroll
      for (int r = 0; r < 8; ++r) pk[r] = tobf((acc[c][r] + bq[mbase + r]) * ATTN_SCALE);
      const size_t bh = (size_t)b * NHEAD + (mbase >> 5);
      *(bf16x8*)(qw + (bh * NPIX + col) * HDIM + (mbase & 31)) = pk;
    } else if (m0 < 2 * MID) {                             // ---- k ----
      const int wr = mbase - MID;
      bf16x8 pk;
#pragma unroll
      for (int r = 0; r < 8; ++r) pk[r] = tobf(acc[c][r] + bkv[wr + r]);
      const size_t bh = (size_t)b * NHEAD + (wr >> 5);
      *(bf16x8*)(kpad + (bh * NKPAD + col) * HDIM + (wr & 31)) = pk;
    } else {                                               // ---- v (transposed) ----
      const int wr = mbase - MID;                          // bkv row
      const int vv = mbase - 2 * MID;                      // v channel
      const size_t bh = (size_t)b * NHEAD + (vv >> 5);
#pragma unroll
      for (int r = 0; r < 8; ++r)
        vT[(bh * HDIM + ((vv + r) & 31)) * NKPAD + col] = tobf(acc[c][r] + bkv[wr + r]);
    }
  }
}

// ---------------------------------------------------------------------------
// Stage 2: flash attention, one wave per (b, head, 16-query tile).
// 32 keys/step: 2 WMMAs for S, shuffle-reduction online softmax, LDS re-layout
// of P (C-frag -> A-frag), 2 WMMAs for P*V. All fragment loads are b128.
// ---------------------------------------------------------------------------
__global__ __launch_bounds__(32) void attn_kernel(
    const __bf16* __restrict__ qw, const __bf16* __restrict__ kpad,
    const __bf16* __restrict__ vT,
    const float* __restrict__ bias_table, const int* __restrict__ rel_index,
    __bf16* __restrict__ ao)
{
  __shared__ __bf16 smP[16][40];   // 16 x 32 P tile; stride 40 keeps 16B alignment

  const int lane = threadIdx.x & 31;
  const int n0   = blockIdx.x * 16;
  const int h    = blockIdx.y;
  const int b    = blockIdx.z;
  const int hi8  = (lane >= 16) ? 8 : 0;
  const int kbB  = (lane >= 16) ? 16 : 0;
  const int l15  = lane & 15;

  const size_t bh = (size_t)b * NHEAD + h;
  const __bf16* qb  = qw   + bh * NPIX  * HDIM;
  const __bf16* kb  = kpad + bh * NKPAD * HDIM;
  const __bf16* vb0 = vT + (bh * HDIM + l15)      * NKPAD;   // d = lane&15
  const __bf16* vb1 = vT + (bh * HDIM + l15 + 16) * NKPAD;   // d = lane&15 + 16

  // Q A-fragment
  const int qn = n0 + l15;
  const v16bf aq = cat16(ldb8(qb + (size_t)qn * HDIM + kbB),
                         ldb8(qb + (size_t)qn * HDIM + kbB + 8));

  v8f oacc0 = {0.f,0.f,0.f,0.f,0.f,0.f,0.f,0.f};
  v8f oacc1 = {0.f,0.f,0.f,0.f,0.f,0.f,0.f,0.f};
  float mrow[8], lrow[8];
#pragma unroll
  for (int r = 0; r < 8; ++r) { mrow[r] = -1e30f; lrow[r] = 0.f; }

  for (int m0 = 0; m0 < NKPAD; m0 += 32) {
    const int c1 = m0 + l15;
    const int c2 = c1 + 16;

    // K^T B-fragments (padding rows are zeroed -> no masks on loads)
    const v16bf bk1 = cat16(ldb8(kb + (size_t)c1 * HDIM + kbB),
                            ldb8(kb + (size_t)c1 * HDIM + kbB + 8));
    const v16bf bk2 = cat16(ldb8(kb + (size_t)c2 * HDIM + kbB),
                            ldb8(kb + (size_t)c2 * HDIM + kbB + 8));
    if (m0 + 32 < NKPAD) {
      __builtin_prefetch(kb + (size_t)(c1 + 32) * HDIM + kbB, 0, 3);
      __builtin_prefetch(vb0 + m0 + 32 + kbB, 0, 3);
    }
    const v8f zro = {0.f,0.f,0.f,0.f,0.f,0.f,0.f,0.f};
    v8f s1 = WMMA_BF16(aq, bk1, zro);
    v8f s2 = WMMA_BF16(aq, bk2, zro);

    // relative-position bias + padding mask
#pragma unroll
    for (int r = 0; r < 8; ++r) {
      const int qr = n0 + r + hi8;
      if (c1 < NPIX) s1[r] += bias_table[(size_t)rel_index[(size_t)qr * NPIX + c1] * NHEAD + h];
      else           s1[r]  = -1e30f;
      if (c2 < NPIX) s2[r] += bias_table[(size_t)rel_index[(size_t)qr * NPIX + c2] * NHEAD + h];
      else           s2[r]  = -1e30f;
    }

    // online softmax: each C-fragment row = one VGPR across a 16-lane half
#pragma unroll
    for (int r = 0; r < 8; ++r) {
      float mx = fmaxf(s1[r], s2[r]);
#pragma unroll
      for (int off = 1; off < 16; off <<= 1) mx = fmaxf(mx, __shfl_xor(mx, off, 32));
      const float mn   = fmaxf(mrow[r], mx);
      const float corr = __expf(mrow[r] - mn);
      const float p1   = __expf(s1[r] - mn);
      const float p2   = __expf(s2[r] - mn);
      float ps = p1 + p2;
#pragma unroll
      for (int off = 1; off < 16; off <<= 1) ps += __shfl_xor(ps, off, 32);
      lrow[r]  = lrow[r] * corr + ps;
      mrow[r]  = mn;
      oacc0[r] *= corr;
      oacc1[r] *= corr;
      s1[r] = p1;
      s2[r] = p2;
    }

    // P: C-layout -> LDS -> A-layout
    __syncthreads();
#pragma unroll
    for (int r = 0; r < 8; ++r) {
      smP[r + hi8][l15]      = tobf(s1[r]);
      smP[r + hi8][16 + l15] = tobf(s2[r]);
    }
    __syncthreads();
    const v16bf ap = cat16(ldb8(&smP[l15][kbB]), ldb8(&smP[l15][kbB + 8]));

    // V B-fragments from transposed V (contiguous along keys)
    const v16bf bv0 = cat16(ldb8(vb0 + m0 + kbB), ldb8(vb0 + m0 + kbB + 8));
    const v16bf bv1 = cat16(ldb8(vb1 + m0 + kbB), ldb8(vb1 + m0 + kbB + 8));
    oacc0 = WMMA_BF16(ap, bv0, oacc0);
    oacc1 = WMMA_BF16(ap, bv1, oacc1);
  }

  // normalize, store to ao [b][n][mid] (mid-contiguous for projection B-frags)
#pragma unroll
  for (int r = 0; r < 8; ++r) {
    const int qr = n0 + r + hi8;
    const float inv = 1.0f / lrow[r];
    __bf16* dst = ao + ((size_t)b * NPIX + qr) * MID + h * HDIM + l15;
    dst[0]  = tobf(oacc0[r] * inv);
    dst[16] = tobf(oacc1[r] * inv);
  }
}

// ---------------------------------------------------------------------------
// Stage 3: output projection. out = gamma * (Wproj (384x384) * AO_b + bproj)
// 64x16 output tile per wave (384 = 6 * 64).
// ---------------------------------------------------------------------------
__global__ __launch_bounds__(64) void proj_kernel(
    const __bf16* __restrict__ ao, const __bf16* __restrict__ wproj,
    const float* __restrict__ bproj, const float* __restrict__ gamma,
    float* __restrict__ out)
{
  const int lane  = threadIdx.x & 31;
  const int wv    = threadIdx.x >> 5;
  const int mtile = blockIdx.y * 2 + wv;      // 0..5 (6 * 64 = 384)
  const int m0    = mtile * 64;
  const int n0    = blockIdx.x * 16;
  const int b     = blockIdx.z;
  const int hi8   = (lane >= 16) ? 8 : 0;
  const int col   = n0 + (lane & 15);
  const int kbA   = (lane >= 16) ? 16 : 0;
  const int kbB   = (lane >= 16) ? 16 : 0;

  const __bf16* wrow0 = wproj + (size_t)(m0 + (lane & 15)) * MID;
  const __bf16* acol  = ao + ((size_t)b * NPIX + col) * MID;

  v8f acc[4];
#pragma unroll
  for (int c = 0; c < 4; ++c) acc[c] = (v8f){0.f,0.f,0.f,0.f,0.f,0.f,0.f,0.f};

  for (int k0 = 0; k0 < MID; k0 += 32) {
    const v16bf bm = cat16(ldb8(acol + k0 + kbB), ldb8(acol + k0 + kbB + 8));
#pragma unroll
    for (int c = 0; c < 4; ++c) {
      const __bf16* wr = wrow0 + (size_t)(16 * c) * MID + k0 + kbA;
      const v16bf a = cat16(ldb8(wr), ldb8(wr + 8));
      acc[c] = WMMA_BF16(a, bm, acc[c]);
    }
  }

#pragma unroll
  for (int c = 0; c < 4; ++c) {
#pragma unroll
    for (int r = 0; r < 8; ++r) {
      const int o = m0 + 16 * c + r + hi8;
      out[((size_t)b * MID + o) * NPIX + col] = (acc[c][r] + bproj[o]) * gamma[o];
    }
  }
}

// ---------------------------------------------------------------------------
extern "C" void kernel_launch(void* const* d_in, const int* in_sizes, int n_in,
                              void* d_out, int out_size, void* d_ws, size_t ws_size,
                              hipStream_t stream)
{
  const float* x          = (const float*)d_in[0];
  const float* Wq         = (const float*)d_in[1];
  const float* bq         = (const float*)d_in[2];
  const float* Wkv        = (const float*)d_in[3];
  const float* bkv        = (const float*)d_in[4];
  const float* bias_table = (const float*)d_in[5];
  const float* Wproj      = (const float*)d_in[6];
  const float* bproj      = (const float*)d_in[7];
  const float* gamma      = (const float*)d_in[8];
  const int*   rel_index  = (const int*)d_in[9];
  float* out = (float*)d_out;

  auto align256 = [](size_t v) { return (v + 255) & ~(size_t)255; };
  char* base = (char*)d_ws;
  size_t off = 0;
  __bf16* wqkv  = (__bf16*)(base + off); off += align256((size_t)3 * MID * CIN * 2);      // 1152x384
  __bf16* wproj = (__bf16*)(base + off); off += align256((size_t)MID * MID * 2);          // 384x384
  __bf16* xT    = (__bf16*)(base + off); off += align256((size_t)NB * NPIX * CIN * 2);
  __bf16* qw    = (__bf16*)(base + off); off += align256((size_t)NB * NHEAD * NPIX  * HDIM * 2);
  __bf16* kpad  = (__bf16*)(base + off); off += align256((size_t)NB * NHEAD * NKPAD * HDIM * 2);
  __bf16* vT    = (__bf16*)(base + off); off += align256((size_t)NB * NHEAD * HDIM * NKPAD * 2);
  __bf16* ao    = (__bf16*)(base + off); off += align256((size_t)NB * NPIX * MID * 2);

  // Prep: weight conversion (Wq then Wkv into one 1152x384 bf16 panel), Wproj,
  // x transpose, k/v padding zero-fill.
  {
    const int nq = MID * CIN, nkv = 2 * MID * CIN, np = MID * MID;
    cvt_kernel<<<dim3((nq  + 255) / 256), 256, 0, stream>>>(Wq,  wqkv,            nq);
    cvt_kernel<<<dim3((nkv + 255) / 256), 256, 0, stream>>>(Wkv, wqkv + (size_t)MID * CIN, nkv);
    cvt_kernel<<<dim3((np  + 255) / 256), 256, 0, stream>>>(Wproj, wproj, np);
  }
  xpose_kernel<<<dim3(49, 24, NB), 256, 0, stream>>>(x, xT);
  pad_zero_kernel<<<dim3((NB * NHEAD * 512 + 255) / 256), 256, 0, stream>>>(kpad, vT);

  // Stage 1: QKV projection — 18 row-tiles (64 rows) x 49 col-tiles per batch
  qkv_kernel<<<dim3(49, 9, NB), 64, 0, stream>>>(xT, wqkv, bq, bkv, qw, kpad, vT);

  // Stage 2: flash attention — one wave per (query-tile, head, batch)
  attn_kernel<<<dim3(49, NHEAD, NB), 32, 0, stream>>>(qw, kpad, vT, bias_table, rel_index, ao);

  // Stage 3: projection + bias + layer scale — 6 row-tiles (64 rows) x 49 col-tiles
  proj_kernel<<<dim3(49, 3, NB), 64, 0, stream>>>(ao, wproj, bproj, gamma, out);
}
